// AttnBias_20246475833912
// MI455X (gfx1250) — compile-verified, matrix operand order checked by reading, not run on hardware
//
#include <hip/hip_runtime.h>
#include <hip/hip_bf16.h>

typedef __attribute__((ext_vector_type(16))) _Float16 v16h;
typedef __attribute__((ext_vector_type(8)))  float    v8f;

#define N_    256
#define NN    65536      // N*N
#define KHOP  32
#define D_    32
#define H_    8
#define WAVES_PER_BLOCK 8

// One wave computes a tile of 16 consecutive (b,n,m) pairs:
//   C[16x16] += A[16x32] (gathered edge features, f16) * Bk[32x16] (hop weights, f16)
// accumulated over the 32 hops, then scaled by 1/path_len and fused with the
// spatial bias gather on the way out.
__global__ __launch_bounds__(256)
void attn_bias_wmma(const int*   __restrict__ spatial_pos,   // [B*N*N]
                    const int*   __restrict__ edge_input,    // [B*N*N*K]
                    const float* __restrict__ spatial_emb,   // [512*H]
                    const float* __restrict__ edge_feat_emb, // [1025*D]
                    const float* __restrict__ edge_pos_emb,  // [K*D*H]
                    float*       __restrict__ out)           // [B*H*N*N]
{
    // B-matrix fragments for all 32 hops, stored in the exact WMMA per-lane
    // register order: [hop][lane][16 halves] = 32 KB.
    __shared__ _Float16 bfrag[KHOP * 32 * 16];

    const int tid  = threadIdx.x;
    const int lane = tid & 31;
    const int wave = tid >> 5;

    // ---- Stage B fragments into LDS (cooperative, once per block) ----
    // B[k][d][n] = edge_pos_emb[k][d*H + n] for n < H, else 0 (pad to 16 cols).
    // Lane L of the wave holds column n=L&15, d-set {0..7,16..23} (L<16)
    // or {8..15,24..31} (L>=16)  -- mirror of the A-matrix K layout.
    for (int f = tid; f < KHOP * 32; f += 256) {
        const int k  = f >> 5;
        const int L  = f & 31;
        const int n  = L & 15;
        const int hi = (L >> 4) * 8;
        _Float16* dst = &bfrag[f * 16];
        if (n < H_) {
            const float* w = edge_pos_emb + k * (D_ * H_) + n;  // stride H_ over d
#pragma unroll
            for (int j = 0; j < 16; ++j) {
                const int d = (j < 8) ? (j + hi) : (j - 8 + 16 + hi);
                dst[j] = (_Float16)w[d * H_];
            }
        } else {
#pragma unroll
            for (int j = 0; j < 16; ++j) dst[j] = (_Float16)0.0f;
        }
    }
    __syncthreads();

    // ---- Each wave owns one 16-pair tile ----
    const int  tile  = blockIdx.x * WAVES_PER_BLOCK + wave;
    const long long base = (long long)tile * 16;   // first flat pair index
    const int  mrow  = lane & 15;                  // pair-in-tile this lane gathers
    const int  dbase = (lane >> 4) * 8;            // 0 (lanes 0-15) / 8 (lanes 16-31)

    const int* eip = edge_input + (base + mrow) * KHOP;

    v8f c = {};
    int cnt = 0;
    for (int k = 0; k < KHOP; ++k) {
        const int idx = eip[k];                    // lane & lane+16 load same value
        cnt += (idx != 0);                         // padding row 0 is all-zero
        // A fragment: d in {dbase..dbase+7} then {dbase+16..dbase+23}
        const float4* row4 = (const float4*)(edge_feat_emb + (long long)idx * D_ + dbase);
        const float4 c0 = row4[0], c1 = row4[1], c2 = row4[4], c3 = row4[5];
        v16h a;
        a[0]  = (_Float16)c0.x; a[1]  = (_Float16)c0.y; a[2]  = (_Float16)c0.z; a[3]  = (_Float16)c0.w;
        a[4]  = (_Float16)c1.x; a[5]  = (_Float16)c1.y; a[6]  = (_Float16)c1.z; a[7]  = (_Float16)c1.w;
        a[8]  = (_Float16)c2.x; a[9]  = (_Float16)c2.y; a[10] = (_Float16)c2.z; a[11] = (_Float16)c2.w;
        a[12] = (_Float16)c3.x; a[13] = (_Float16)c3.y; a[14] = (_Float16)c3.z; a[15] = (_Float16)c3.w;

        const v16h b = *(const v16h*)&bfrag[(k * 32 + lane) * 16];  // 2x ds_load_b128

        c = __builtin_amdgcn_wmma_f32_16x16x32_f16(
                /*neg_a=*/false, a, /*neg_b=*/false, b,
                /*c_mod=*/(short)0, c, /*reuse_a=*/false, /*reuse_b=*/false);
    }

    // ---- Epilogue: C[lane, r] is (M = r + 8*(lane>=16), h = lane&15) ----
    const int   h    = lane & 15;
    const int   mtop = (lane >> 4) * 8;
    const float cntf = (float)cnt;

    // Redistribute path-length counts while EXEC is still all-ones.
    float pls[8];
#pragma unroll
    for (int r = 0; r < 8; ++r)
        pls[r] = fmaxf(__shfl(cntf, mtop + r, 32), 1.0f);

    if (h < H_) {
#pragma unroll
        for (int r = 0; r < 8; ++r) {
            const long long p   = base + mtop + r;     // flat (b,n,m) pair
            const int       sp  = spatial_pos[p];
            const float     sb  = spatial_emb[sp * H_ + h];
            const long long bi  = p >> 16;             // p / NN
            const long long nm  = p & 0xFFFF;          // p % NN
            out[bi * (long long)(H_ * NN) + (long long)h * NN + nm] =
                sb + c[r] / pls[r];
        }
    }
}

extern "C" void kernel_launch(void* const* d_in, const int* in_sizes, int n_in,
                              void* d_out, int out_size, void* d_ws, size_t ws_size,
                              hipStream_t stream) {
    const int*   spatial_pos   = (const int*)d_in[0];
    const int*   edge_input    = (const int*)d_in[1];
    const float* spatial_emb   = (const float*)d_in[2];
    const float* edge_feat_emb = (const float*)d_in[3];
    const float* edge_pos_emb  = (const float*)d_in[4];
    float*       out           = (float*)d_out;

    const int pairs  = in_sizes[0];                    // B*N*N = 262144
    const int tiles  = pairs / 16;                     // 16384
    const int blocks = tiles / WAVES_PER_BLOCK;        // 2048

    hipLaunchKernelGGL(attn_bias_wmma, dim3(blocks), dim3(256), 0, stream,
                       spatial_pos, edge_input, spatial_emb, edge_feat_emb,
                       edge_pos_emb, out);
}
